// Attention_33492154974312
// MI455X (gfx1250) — compile-verified, matrix operand order checked by reading, not run on hardware
//
#include <hip/hip_runtime.h>
#include <hip/hip_bf16.h>

typedef __bf16 bf16_t;
typedef __bf16 v16bf __attribute__((ext_vector_type(16)));
typedef float  v8f   __attribute__((ext_vector_type(8)));
typedef unsigned int u32;
typedef u32 u32x4 __attribute__((ext_vector_type(4)));
typedef int  i32x4 __attribute__((ext_vector_type(4)));
typedef int  i32x8 __attribute__((ext_vector_type(8)));

#define B_   256
#define N_   198
#define C_   768
#define H_   12
#define DH_  64
#define NPR_ 208   // padded rows (13*16)
#define NPC_ 224   // padded cols (7*32)
#define ROWS_ (B_*N_)     // 50688
#define BH_  (B_*H_)      // 3072

#if defined(__has_builtin)
#  if __has_builtin(__builtin_amdgcn_tensor_load_to_lds) && __has_builtin(__builtin_amdgcn_s_wait_tensorcnt)
#    define USE_TDM 1
#  endif
#endif
#ifndef USE_TDM
#  define USE_TDM 0
#endif

// ---------------- WMMA fragment helpers (wave32, 16x16x32 bf16) ----------------
// A-frag (16x32 MxK) and B-frag (32x16 KxN given as Bt[N][K] row-major):
//   lane<16 : row (lane), K = k0+0..7 (elems 0..7) and k0+16..23 (elems 8..15)
//   lane>=16: row (lane-16), K = k0+8..15 and k0+24..31
__device__ inline v16bf load_frag(const bf16_t* base, int lane, int ld, int k0) {
    const bf16_t* p = base + (size_t)(lane & 15) * ld + k0 + ((lane >> 4) << 3);
    union { v16bf v; uint4 u[2]; } f;
    f.u[0] = *(const uint4*)(p);
    f.u[1] = *(const uint4*)(p + 16);
    return f.v;
}

__device__ inline v8f wmma_bf16(v16bf a, v16bf b, v8f c) {
    return __builtin_amdgcn_wmma_f32_16x16x32_bf16(false, a, false, b, (short)0, c, false, false);
}

__device__ inline float wave_max(float v) {
    #pragma unroll
    for (int m = 16; m > 0; m >>= 1) v = fmaxf(v, __shfl_xor(v, m, 32));
    return v;
}
__device__ inline float wave_sum(float v) {
    #pragma unroll
    for (int m = 16; m > 0; m >>= 1) v += __shfl_xor(v, m, 32);
    return v;
}
__device__ inline float sigmoidf_(float x) { return 1.0f / (1.0f + __expf(-x)); }

// ---------------- TDM 2D tile -> LDS (D# per CDNA5 ISA ch.8) ----------------
#if USE_TDM
// Load a (rows x 32) bf16 tile (row stride ld elements) to lds_byte_off.
__device__ inline void tdm_load_2d(u32 lds_byte_off, const bf16_t* g, u32 rows, u32 ld) {
    unsigned long long ga = (unsigned long long)g;
    u32x4 g0;
    g0[0] = 1u;                                            // count=1, user descriptor
    g0[1] = lds_byte_off;                                  // lds_addr
    g0[2] = (u32)(ga & 0xffffffffu);                       // global_addr[31:0]
    g0[3] = (u32)((ga >> 32) & 0x01ffffffu) | (2u << 30);  // global_addr[56:32] | type=2
    i32x8 g1;
    g1[0] = (int)(1u << 16);                               // data_size=1 (2 bytes)
    g1[1] = (int)(32u << 16);                              // tensor_dim0[15:0]=32 @bits63:48
    g1[2] = (int)((rows & 0xffffu) << 16);                 // tensor_dim1[15:0] @bits95:80
    g1[3] = (int)(32u << 16);                              // tile_dim0=32 @bits127:112
    g1[4] = (int)(rows & 0xffffu);                         // tile_dim1=rows
    g1[5] = (int)ld;                                       // tensor_dim0_stride[31:0]
    g1[6] = 0;
    g1[7] = 0;
    i32x4 z4 = {0, 0, 0, 0};
    i32x8 z8 = {0, 0, 0, 0, 0, 0, 0, 0};
    __builtin_amdgcn_tensor_load_to_lds(g0, g1, z4, z4, z8, 0);
}
__device__ inline u32 lds_off(const void* p) { return (u32)(unsigned long long)p; }
#endif

// One K-step of the 32x32 register block: 4 LDS fragment loads, 4 WMMAs.
__device__ inline void mma_step(const bf16_t* Ab, const bf16_t* Bb, int lane,
                                int mrow, int ncol, v8f acc[2][2]) {
    v16bf fa0 = load_frag(Ab + (size_t)(mrow)      * 32, lane, 32, 0);
    v16bf fa1 = load_frag(Ab + (size_t)(mrow + 16) * 32, lane, 32, 0);
    v16bf fb0 = load_frag(Bb + (size_t)(ncol)      * 32, lane, 32, 0);
    v16bf fb1 = load_frag(Bb + (size_t)(ncol + 16) * 32, lane, 32, 0);
    acc[0][0] = wmma_bf16(fa0, fb0, acc[0][0]);
    acc[0][1] = wmma_bf16(fa0, fb1, acc[0][1]);
    acc[1][0] = wmma_bf16(fa1, fb0, acc[1][0]);
    acc[1][1] = wmma_bf16(fa1, fb1, acc[1][1]);
}

#if !USE_TDM
// Fallback staging: plain global->LDS copies.
__device__ inline void stage_fallback(bf16_t* Abuf, bf16_t* Bbuf,
                                      const bf16_t* Ag, const bf16_t* Bg,
                                      int ld, int k0, int t) {
    #pragma unroll
    for (int i = 0; i < 2; i++) {
        int ch = t * 2 + i;                 // 512 chunks of 8 elems: 128x32
        int r = ch >> 2, cc = (ch & 3) * 8;
        *(uint4*)(Abuf + ch * 8) = *(const uint4*)(Ag + (size_t)r * ld + k0 + cc);
    }
    {
        int r = t >> 2, cc = (t & 3) * 8;   // 256 chunks: 64x32
        *(uint4*)(Bbuf + t * 8) = *(const uint4*)(Bg + (size_t)r * ld + k0 + cc);
    }
}
#endif

// ---------------- 1) casts / transposes ----------------
__global__ __launch_bounds__(256) void cast_x_kernel(const float* __restrict__ in,
                                                     bf16_t* __restrict__ out, int n) {
    int i = blockIdx.x * 256 + threadIdx.x;
    if (i < n) out[i] = (bf16_t)in[i];
}

// out[n][k] = in[k][n]; out has K columns, in has Nc columns
__global__ __launch_bounds__(256) void transpose_cast_kernel(const float* __restrict__ in,
                                                             bf16_t* __restrict__ out,
                                                             int K, int Nc) {
    int i = blockIdx.x * 256 + threadIdx.x;
    if (i < K * Nc) {
        int n = i / K, k = i % K;
        out[i] = (bf16_t)in[(size_t)k * Nc + n];
    }
}

// ---------------- 2) qkv GEMM: [50688,768] x [768,2304], TDM-staged ----------------
__global__ __launch_bounds__(256) void qkv_gemm(const bf16_t* __restrict__ xb,
                                                const bf16_t* __restrict__ wT,
                                                bf16_t* __restrict__ q,
                                                bf16_t* __restrict__ k,
                                                bf16_t* __restrict__ vT) {
    __shared__ bf16_t Abuf[2][128 * 32];
    __shared__ bf16_t Bbuf[2][64 * 32];
    const int NTB = (3 * C_) / 64;                 // 36
    int tm = blockIdx.x / NTB, tn = blockIdx.x % NTB;
    int wave = threadIdx.x >> 5, lane = threadIdx.x & 31;
    int mrow = (wave >> 1) * 32, ncol = (wave & 1) * 32;
    const bf16_t* Ag = xb + (size_t)tm * 128 * C_;
    const bf16_t* Bg = wT + (size_t)tn * 64 * C_;
    v8f acc[2][2];
    #pragma unroll
    for (int i = 0; i < 2; i++)
        #pragma unroll
        for (int j = 0; j < 2; j++) acc[i][j] = v8f{0.f, 0.f, 0.f, 0.f, 0.f, 0.f, 0.f, 0.f};
    const int KS = C_ / 32;                        // 24

#if USE_TDM
    if (wave == 0) {
        tdm_load_2d(lds_off(&Abuf[0][0]), Ag, 128, C_);
        tdm_load_2d(lds_off(&Bbuf[0][0]), Bg, 64, C_);
    }
    for (int it = 0; it < KS; ++it) {
        int cur = it & 1;
        if (wave == 0) {
            if (it + 1 < KS) {
                tdm_load_2d(lds_off(&Abuf[cur ^ 1][0]), Ag + (it + 1) * 32, 128, C_);
                tdm_load_2d(lds_off(&Bbuf[cur ^ 1][0]), Bg + (it + 1) * 32, 64, C_);
                __builtin_amdgcn_s_wait_tensorcnt(2);
            } else {
                __builtin_amdgcn_s_wait_tensorcnt(0);
            }
        }
        __syncthreads();
        mma_step(&Abuf[cur][0], &Bbuf[cur][0], lane, mrow, ncol, acc);
        __syncthreads();
    }
#else
    for (int it = 0; it < KS; ++it) {
        stage_fallback(&Abuf[0][0], &Bbuf[0][0], Ag, Bg, C_, it * 32, threadIdx.x);
        __syncthreads();
        mma_step(&Abuf[0][0], &Bbuf[0][0], lane, mrow, ncol, acc);
        __syncthreads();
    }
#endif

    int nl = lane & 15, mb = (lane >> 4) * 8;
    #pragma unroll
    for (int sm = 0; sm < 2; sm++)
        #pragma unroll
        for (int sn = 0; sn < 2; sn++) {
            int c = tn * 64 + ncol + sn * 16 + nl;
            int which = c / C_, rem = c % C_, h = rem >> 6, d = rem & 63;
            #pragma unroll
            for (int e = 0; e < 8; e++) {
                int row = tm * 128 + mrow + sm * 16 + mb + e;
                int b = row / N_, n = row % N_;
                bf16_t val = (bf16_t)acc[sm][sn][e];
                size_t bh = (size_t)b * H_ + h;
                if (which == 0)      q[(bh * NPR_ + n) * DH_ + d] = val;
                else if (which == 1) k[(bh * NPR_ + n) * DH_ + d] = val;
                else                 vT[(bh * DH_ + d) * NPC_ + n] = val;
            }
        }
}

// ---------------- 3) q0 heads -> D, cen, mgb ----------------
__global__ __launch_bounds__(256) void mgb_kernel(const bf16_t* __restrict__ q,
                                                  const float* __restrict__ wD,
                                                  const float* __restrict__ bD,
                                                  const float* __restrict__ wcen,
                                                  const float* __restrict__ bcen,
                                                  float* __restrict__ out_cen,
                                                  float* __restrict__ mgb) {
    __shared__ float part[9][256];
    __shared__ float sig2[3], cox[3], coy[3];
    __shared__ float mj[3][196];
    __shared__ float msum[3];
    int b = blockIdx.x, t = threadIdx.x;

    float p[9];
    #pragma unroll
    for (int j = 0; j < 9; j++) p[j] = 0.f;
    for (int c = t; c < C_; c += 256) {
        int h = c >> 6, d = c & 63;
        float qv = (float)q[(((size_t)b * H_ + h) * NPR_) * DH_ + d];
        #pragma unroll
        for (int j = 0; j < 3; j++) p[j]     += qv * wD[c * 3 + j];
        #pragma unroll
        for (int j = 0; j < 6; j++) p[3 + j] += qv * wcen[c * 6 + j];
    }
    #pragma unroll
    for (int j = 0; j < 9; j++) part[j][t] = p[j];
    __syncthreads();
    for (int s = 128; s > 0; s >>= 1) {
        if (t < s)
            #pragma unroll
            for (int j = 0; j < 9; j++) part[j][t] += part[j][t + s];
        __syncthreads();
    }
    if (t == 0) {
        #pragma unroll
        for (int j = 0; j < 3; j++) {
            float Dv = 224.0f * sigmoidf_(part[j][0] + bD[j]);
            float s  = Dv * 0.5f;
            sig2[j] = s * s;
            float cx = sigmoidf_(part[3 + 2 * j][0]     + bcen[2 * j]);
            float cy = sigmoidf_(part[3 + 2 * j + 1][0] + bcen[2 * j + 1]);
            out_cen[b * 6 + 2 * j]     = cx;
            out_cen[b * 6 + 2 * j + 1] = cy;
            cox[j] = cx * 223.0f;
            coy[j] = cy * 223.0f;
        }
    }
    __syncthreads();
    if (t < 196) {
        float gx = 8.0f + 16.0f * (float)(t / 14);
        float gy = 8.0f + 16.0f * (float)(t % 14);
        #pragma unroll
        for (int j = 0; j < 3; j++) {
            float dx = cox[j] - gx, dy = coy[j] - gy;
            mj[j][t] = __expf(-(dx * dx + dy * dy) / (2.0f * sig2[j])) / (2.0f * 3.14f * sig2[j]);
        }
    }
    __syncthreads();
    #pragma unroll
    for (int j = 0; j < 3; j++) part[j][t] = (t < 196) ? mj[j][t] : 0.f;
    __syncthreads();
    for (int s = 128; s > 0; s >>= 1) {
        if (t < s)
            #pragma unroll
            for (int j = 0; j < 3; j++) part[j][t] += part[j][t + s];
        __syncthreads();
    }
    if (t < 3) msum[t] = part[t][0];
    __syncthreads();
    if (t < 196) {
        float g = 0.f;
        #pragma unroll
        for (int j = 0; j < 3; j++) g += mj[j][t] / msum[j];
        mgb[(size_t)b * N_ + 2 + t] = g * (1.0f / 3.0f);
    }
    if (t < 2) mgb[(size_t)b * N_ + t] = 1.0f;
}

// ---------------- 4) scores + softmax per (b,h,row-band) ----------------
__global__ __launch_bounds__(32) void attn_softmax(const bf16_t* __restrict__ q,
                                                   const bf16_t* __restrict__ kk,
                                                   bf16_t* __restrict__ attn,
                                                   float* __restrict__ cls) {
    __shared__ float srow[16][NPC_];
    int lane = threadIdx.x;
    int rt = blockIdx.x, h = blockIdx.y, b = blockIdx.z;
    size_t bh = (size_t)b * H_ + h;
    const bf16_t* Aq = q  + (bh * NPR_ + (size_t)rt * 16) * DH_;
    const bf16_t* Bk = kk + bh * NPR_ * DH_;

    v16bf fa0 = load_frag(Aq, lane, DH_, 0);
    v16bf fa1 = load_frag(Aq, lane, DH_, 32);
    int nl = lane & 15, mb = (lane >> 4) * 8;
    for (int j = 0; j < 13; j++) {
        v8f a = {0.f, 0.f, 0.f, 0.f, 0.f, 0.f, 0.f, 0.f};
        const bf16_t* Bj = Bk + (size_t)j * 16 * DH_;
        a = wmma_bf16(fa0, load_frag(Bj, lane, DH_, 0), a);
        a = wmma_bf16(fa1, load_frag(Bj, lane, DH_, 32), a);
        #pragma unroll
        for (int e = 0; e < 8; e++) srow[mb + e][j * 16 + nl] = a[e] * 0.125f;
    }
    __syncthreads();

    for (int r = 0; r < 16; r++) {
        int n = rt * 16 + r;
        float mx = -1e30f;
        for (int c = lane; c < N_; c += 32) mx = fmaxf(mx, srow[r][c]);
        mx = wave_max(mx);
        float sum = 0.f;
        for (int c = lane; c < N_; c += 32) {
            float e = __expf(srow[r][c] - mx);
            srow[r][c] = e;
            sum += e;
        }
        sum = wave_sum(sum);
        float inv = 1.0f / sum;
        if (n < N_) {
            bf16_t* dst = attn + (bh * NPR_ + n) * NPC_;
            for (int c = lane; c < NPC_; c += 32) {
                float pv = (c < N_) ? srow[r][c] * inv : 0.0f;
                dst[c] = (bf16_t)pv;
                if (n == 0 && c < N_) cls[bh * N_ + c] = pv;
            }
        }
    }
}

// ---------------- 5) hs = cls * mgb, normalized, stored [H,B,N] ----------------
__global__ __launch_bounds__(32) void hs_kernel(const float* __restrict__ cls,
                                                const float* __restrict__ mgb,
                                                float* __restrict__ hsbuf) {
    int lane = threadIdx.x;
    int h = blockIdx.x, b = blockIdx.y;
    const float* crow = cls + ((size_t)b * H_ + h) * N_;
    const float* mrow = mgb + (size_t)b * N_;
    float sum = 0.f;
    for (int n = lane; n < N_; n += 32) sum += crow[n] * mrow[n];
    sum = wave_sum(sum);
    float inv = 1.0f / sum;
    float* drow = hsbuf + ((size_t)h * B_ + b) * N_;
    for (int n = lane; n < N_; n += 32) drow[n] = crow[n] * mrow[n] * inv;
}

// ---------------- 6) masked_scatter flat trick: new_row flat == hs flat ----------------
__global__ __launch_bounds__(256) void scatter_newrow(const float* __restrict__ hsbuf,
                                                      float* __restrict__ out_cls,
                                                      bf16_t* __restrict__ attn) {
    long i = (long)blockIdx.x * 256 + threadIdx.x;
    if (i >= (long)B_ * H_ * N_) return;
    float v = hsbuf[i];            // hs stored [H,B,N]; reading flat == reshape(-1)
    out_cls[i] = v;                // out index interpreted [B,H,N] flat == i
    long b = i / (H_ * N_);
    int rem = (int)(i % (H_ * N_));
    int h = rem / N_, n = rem % N_;
    attn[(((size_t)b * H_ + h) * NPR_ + 0) * NPC_ + n] = (bf16_t)v;
}

// ---------------- 7) out = attn @ v  (K padded to 224, A zeros kill pad) ----------------
__global__ __launch_bounds__(256) void av_gemm(const bf16_t* __restrict__ attn,
                                               const bf16_t* __restrict__ vT,
                                               bf16_t* __restrict__ outA) {
    int wave = threadIdx.x >> 5, lane = threadIdx.x & 31;
    long tile = (long)blockIdx.x * 8 + wave;
    if (tile >= (long)BH_ * 13 * 4) return;
    int ct = (int)(tile & 3);
    long r1 = tile >> 2;
    int rt = (int)(r1 % 13);
    long bh = r1 / 13;
    const bf16_t* A  = attn + ((size_t)bh * NPR_ + (size_t)rt * 16) * NPC_;
    const bf16_t* Bt = vT   + ((size_t)bh * DH_ + (size_t)ct * 16) * NPC_;
    v8f acc = {0.f, 0.f, 0.f, 0.f, 0.f, 0.f, 0.f, 0.f};
    for (int k0 = 0; k0 < NPC_; k0 += 32)
        acc = wmma_bf16(load_frag(A, lane, NPC_, k0), load_frag(Bt, lane, NPC_, k0), acc);

    int nl = lane & 15, mb = (lane >> 4) * 8;
    int b = (int)(bh / H_), h = (int)(bh % H_);
    int d = ct * 16 + nl;
    #pragma unroll
    for (int e = 0; e < 8; e++) {
        int n = rt * 16 + mb + e;
        if (n < N_) outA[((size_t)b * N_ + n) * C_ + h * DH_ + d] = (bf16_t)acc[e];
    }
}

// ---------------- 8) proj: [50688,768] x [768,768] + bias, TDM-staged ----------------
__global__ __launch_bounds__(256) void proj_gemm(const bf16_t* __restrict__ A0,
                                                 const bf16_t* __restrict__ wT,
                                                 const float* __restrict__ bias,
                                                 float* __restrict__ out) {
    __shared__ bf16_t Abuf[2][128 * 32];
    __shared__ bf16_t Bbuf[2][64 * 32];
    const int NTB = C_ / 64;                       // 12
    int tm = blockIdx.x / NTB, tn = blockIdx.x % NTB;
    int wave = threadIdx.x >> 5, lane = threadIdx.x & 31;
    int mrow = (wave >> 1) * 32, ncol = (wave & 1) * 32;
    const bf16_t* Ag = A0 + (size_t)tm * 128 * C_;
    const bf16_t* Bg = wT + (size_t)tn * 64 * C_;
    v8f acc[2][2];
    #pragma unroll
    for (int i = 0; i < 2; i++)
        #pragma unroll
        for (int j = 0; j < 2; j++) acc[i][j] = v8f{0.f, 0.f, 0.f, 0.f, 0.f, 0.f, 0.f, 0.f};
    const int KS = C_ / 32;                        // 24

#if USE_TDM
    if (wave == 0) {
        tdm_load_2d(lds_off(&Abuf[0][0]), Ag, 128, C_);
        tdm_load_2d(lds_off(&Bbuf[0][0]), Bg, 64, C_);
    }
    for (int it = 0; it < KS; ++it) {
        int cur = it & 1;
        if (wave == 0) {
            if (it + 1 < KS) {
                tdm_load_2d(lds_off(&Abuf[cur ^ 1][0]), Ag + (it + 1) * 32, 128, C_);
                tdm_load_2d(lds_off(&Bbuf[cur ^ 1][0]), Bg + (it + 1) * 32, 64, C_);
                __builtin_amdgcn_s_wait_tensorcnt(2);
            } else {
                __builtin_amdgcn_s_wait_tensorcnt(0);
            }
        }
        __syncthreads();
        mma_step(&Abuf[cur][0], &Bbuf[cur][0], lane, mrow, ncol, acc);
        __syncthreads();
    }
#else
    for (int it = 0; it < KS; ++it) {
        stage_fallback(&Abuf[0][0], &Bbuf[0][0], Ag, Bg, C_, it * 32, threadIdx.x);
        __syncthreads();
        mma_step(&Abuf[0][0], &Bbuf[0][0], lane, mrow, ncol, acc);
        __syncthreads();
    }
#endif

    int nl = lane & 15, mb = (lane >> 4) * 8;
    #pragma unroll
    for (int sm = 0; sm < 2; sm++)
        #pragma unroll
        for (int sn = 0; sn < 2; sn++) {
            int c = tn * 64 + ncol + sn * 16 + nl;
            float bv = bias[c];
            #pragma unroll
            for (int e = 0; e < 8; e++) {
                int row = tm * 128 + mrow + sm * 16 + mb + e;
                out[(size_t)row * C_ + c] = acc[sm][sn][e] + bv;
            }
        }
}

extern "C" void kernel_launch(void* const* d_in, const int* in_sizes, int n_in,
                              void* d_out, int out_size, void* d_ws, size_t ws_size,
                              hipStream_t stream) {
    (void)in_sizes; (void)n_in; (void)out_size; (void)ws_size;
    const float* x      = (const float*)d_in[0];
    const float* w_qkv  = (const float*)d_in[1];
    const float* w_proj = (const float*)d_in[2];
    const float* b_proj = (const float*)d_in[3];
    const float* w_D    = (const float*)d_in[4];
    const float* b_D    = (const float*)d_in[5];
    const float* w_cen  = (const float*)d_in[6];
    const float* b_cen  = (const float*)d_in[7];

    float* out      = (float*)d_out;
    float* out_cen  = out;                                // [256,3,2]
    float* out_cls  = out + 1536;                         // [256,12,198]
    float* out_proj = out + 1536 + (size_t)B_ * H_ * N_;  // [256,198,768]

    char* ws = (char*)d_ws;
    size_t off = 0;
    auto alloc = [&](size_t bytes) -> char* {
        char* p = ws + off;
        off += (bytes + 255) & ~(size_t)255;
        return p;
    };
    bf16_t* xb     = (bf16_t*)alloc((size_t)ROWS_ * C_ * 2);
    bf16_t* wqkvT  = (bf16_t*)alloc((size_t)3 * C_ * C_ * 2);
    bf16_t* wprojT = (bf16_t*)alloc((size_t)C_ * C_ * 2);
    bf16_t* qb     = (bf16_t*)alloc((size_t)BH_ * NPR_ * DH_ * 2);
    bf16_t* kb     = (bf16_t*)alloc((size_t)BH_ * NPR_ * DH_ * 2);
    bf16_t* vT     = (bf16_t*)alloc((size_t)BH_ * DH_ * NPC_ * 2);
    bf16_t* attn   = (bf16_t*)alloc((size_t)BH_ * NPR_ * NPC_ * 2);
    float*  cls    = (float*)alloc((size_t)BH_ * N_ * 4);
    float*  hsbuf  = (float*)alloc((size_t)BH_ * N_ * 4);
    float*  mgb    = (float*)alloc((size_t)B_ * N_ * 4);
    bf16_t* outA   = xb;  // xb dead after qkv_gemm; reuse for attn@v result

    const int nX = ROWS_ * C_;  // 38,928,384
    cast_x_kernel<<<(nX + 255) / 256, 256, 0, stream>>>(x, xb, nX);
    transpose_cast_kernel<<<(3 * C_ * C_ + 255) / 256, 256, 0, stream>>>(w_qkv, wqkvT, C_, 3 * C_);
    transpose_cast_kernel<<<(C_ * C_ + 255) / 256, 256, 0, stream>>>(w_proj, wprojT, C_, C_);

    // qkv GEMM: 396 x 36 macro-tiles of 128x64
    qkv_gemm<<<(ROWS_ / 128) * ((3 * C_) / 64), 256, 0, stream>>>(xb, wqkvT, qb, kb, vT);

    mgb_kernel<<<B_, 256, 0, stream>>>(qb, w_D, b_D, w_cen, b_cen, out_cen, mgb);

    attn_softmax<<<dim3(13, H_, B_), 32, 0, stream>>>(qb, kb, attn, cls);

    hs_kernel<<<dim3(H_, B_), 32, 0, stream>>>(cls, mgb, hsbuf);

    {
        long n = (long)B_ * H_ * N_;  // 608,256
        scatter_newrow<<<(unsigned)((n + 255) / 256), 256, 0, stream>>>(hsbuf, out_cls, attn);
    }

    {   // attn@v: 3072 * 13 * 4 tiles
        long tiles = (long)BH_ * 13 * 4;
        av_gemm<<<(unsigned)((tiles + 7) / 8), 256, 0, stream>>>(attn, vT, outA);
    }

    // proj: 396 x 12 macro-tiles of 128x64
    proj_gemm<<<(ROWS_ / 128) * (C_ / 64), 256, 0, stream>>>(outA, wprojT, b_proj, out_proj);
}
